// DeepFM_12171937317245
// MI455X (gfx1250) — matrix operand order repaired and checked by reading, not compiled, not run on hardware
//
#include <hip/hip_runtime.h>
#include <hip/hip_bf16.h>

typedef __attribute__((ext_vector_type(16))) _Float16 v16h;
typedef __attribute__((ext_vector_type(8)))  _Float16 v8h;
typedef __attribute__((ext_vector_type(4)))  _Float16 v4h;
typedef __attribute__((ext_vector_type(8)))  float    v8f;
typedef __attribute__((ext_vector_type(4)))  float    v4f;

#define MT        64        // batch rows per block
#define NTHR      256       // 8 waves (wave32): 2 row-tiles x 4 col-tiles per wave
#define FEAT      64
#define NUM_DISC  62
#define DNN_IN    1024
#define H1N       256
#define H2N       128
#define OUTW      129

// padded LDS strides (in halves) -> conflict-free row access mod 64 banks
#define ESTRIDE   (DNN_IN + 8)   // 516 dwords, %64 = 4
#define WBSTRIDE  40             // 20 dwords, 16 rows -> distinct banks
#define H1STRIDE  (H1N + 8)      // 132 dwords, %64 = 4
#define W2STRIDE  (H1N + 8)

__device__ __forceinline__ v8h cvt8(const v4f a, const v4f b) {
    v8h r;
#pragma unroll
    for (int i = 0; i < 4; ++i) { r[i] = (_Float16)a[i]; r[i + 4] = (_Float16)b[i]; }
    return r;
}

// A-fragment (16x32 f16 MxK): lane -> row M=lane&15; contiguous K from k0+16*hi
__device__ __forceinline__ v16h lds_fragA(const _Float16* p) {
    v8h a = *(const v8h*)(p);
    v8h b = *(const v8h*)(p + 8);
    v16h r;
#pragma unroll
    for (int i = 0; i < 8; ++i) { r[i] = a[i]; r[i + 8] = b[i]; }
    return r;
}

// B-fragment (32x16 f16 KxN): lane holds column N=lane&15, contiguous K from k0+16*hi
__device__ __forceinline__ v16h lds_fragB(const _Float16* p) {
    v8h a = *(const v8h*)(p);
    v8h b = *(const v8h*)(p + 8);
    v16h r;
#pragma unroll
    for (int i = 0; i < 8; ++i) { r[i] = a[i]; r[i + 8] = b[i]; }
    return r;
}

__device__ __forceinline__ v8f wmma_f16(const v16h a, const v16h b, const v8f c) {
    return __builtin_amdgcn_wmma_f32_16x16x32_f16(false, a, false, b, (short)0, c,
                                                  false, false);
}

// CDNA5 async global->LDS copy (ASYNCcnt-tracked). Generic-pointer low 32 bits are the
// wave-relative LDS address.
__device__ __forceinline__ void async_ld_b128(const _Float16* gp, const _Float16* lp) {
    uint32_t l = (uint32_t)(uintptr_t)lp;
    asm volatile("global_load_async_to_lds_b128 %0, %1, off"
                 :: "v"(l), "v"(gp) : "memory");
}
__device__ __forceinline__ void wait_async0() {
    asm volatile("s_wait_asynccnt 0x0" ::: "memory");
}

// prologue: W1 (256x1024 f32) -> f16 in workspace so the hot loop streams half the bytes
__global__ __launch_bounds__(256) void w1_to_f16(const float* __restrict__ W1,
                                                 _Float16* __restrict__ o) {
    const int i = (blockIdx.x * 256 + threadIdx.x) * 4;
    v4f a = *(const v4f*)(W1 + i);
    v4h h;
#pragma unroll
    for (int d = 0; d < 4; ++d) h[d] = (_Float16)a[d];
    *(v4h*)(o + i) = h;
}

template <bool USE_ASYNC>
__global__ __launch_bounds__(NTHR, 1)
void deepfm_fused(const float* __restrict__ x,   const float* __restrict__ emb,
                  const float* __restrict__ Wc,  const float* __restrict__ bc,
                  const float* __restrict__ Wf,  const float* __restrict__ bfp,
                  const float* __restrict__ W1,  const float* __restrict__ b1,
                  const float* __restrict__ W2,  const float* __restrict__ b2,
                  const _Float16* __restrict__ wsW1,
                  float* __restrict__ out)
{
    __shared__ _Float16 Esh [MT * ESTRIDE];          // 132096 B : f16 embeddings (A of GEMM1)
    __shared__ _Float16 W2sh[H2N * W2STRIDE];        //  67584 B : f16 W2 (resident)
    __shared__ _Float16 H1sh[MT * H1STRIDE];         //  33792 B : relu(layer1) f16
    __shared__ _Float16 Wbuf[2][H1N * WBSTRIDE];     //  40960 B : double-buffered W1 chunk
    __shared__ float    fmf [MT];                    //    256 B

    const int tid  = threadIdx.x;
    const int wv   = tid >> 5;
    const int lane = tid & 31;
    const int lhi  = lane >> 4;
    const int lm   = lane & 15;
    const int row0 = blockIdx.x * MT;

    // ---------------- Phase A: zero fm accum, stage W2 -> f16 LDS ----------------
    if (tid < MT) fmf[tid] = 0.0f;
    {
        const int row = tid >> 1;              // 0..127
        const int seg = (tid & 1) * 128;       // 128-col half
        const float* src = W2 + (size_t)row * H1N + seg;
        _Float16*    dst = W2sh + row * W2STRIDE + seg;
#pragma unroll
        for (int i = 0; i < 8; ++i) {
            v4f a0 = *(const v4f*)(src + i * 16 + 0);
            v4f a1 = *(const v4f*)(src + i * 16 + 4);
            v4f a2 = *(const v4f*)(src + i * 16 + 8);
            v4f a3 = *(const v4f*)(src + i * 16 + 12);
            *(v8h*)(dst + i * 16 + 0) = cvt8(a0, a1);
            *(v8h*)(dst + i * 16 + 8) = cvt8(a2, a3);
        }
    }
    __syncthreads();

    // ---------------- Phase B: build embeddings (f16) + fm_first partials ----------------
#pragma unroll
    for (int i = 0; i < (MT * FEAT) / NTHR; ++i) {
        const int task = i * NTHR + tid;
        const int r = task >> 6;
        const int f = task & 63;
        const float* xrow = x + (size_t)(row0 + r) * FEAT;
        const float xv = xrow[f];
        float e[16];
        if (f < NUM_DISC) {
            const int idx = (xv != 0.0f) ? 1 : 0;
            const float* ep = emb + ((size_t)(f * 2 + idx) << 4);
            v4f a0 = *(const v4f*)(ep + 0);
            v4f a1 = *(const v4f*)(ep + 4);
            v4f a2 = *(const v4f*)(ep + 8);
            v4f a3 = *(const v4f*)(ep + 12);
#pragma unroll
            for (int d = 0; d < 4; ++d) {
                e[d] = a0[d]; e[d + 4] = a1[d]; e[d + 8] = a2[d]; e[d + 12] = a3[d];
            }
        } else {
            const int c = f - NUM_DISC;
            const float x62 = xrow[62], x63 = xrow[63];
#pragma unroll
            for (int d = 0; d < 16; ++d) {
                const int j = c * 16 + d;                  // Wc is (32,2) row-major
                e[d] = fmaf(x62, Wc[2 * j], fmaf(x63, Wc[2 * j + 1], bc[j]));
            }
        }
        _Float16* dst = Esh + r * ESTRIDE + f * 16;
        v8h h0, h1;
#pragma unroll
        for (int d = 0; d < 8; ++d) { h0[d] = (_Float16)e[d]; h1[d] = (_Float16)e[d + 8]; }
        *(v8h*)(dst)     = h0;
        *(v8h*)(dst + 8) = h1;
        atomicAdd(&fmf[r], xv * Wf[f]);                    // ds_add_f32
    }
    __syncthreads();

    // ---------------- Phase B2: FM second-order + output column 0 ----------------
    if (tid < MT) {
        const _Float16* er = Esh + tid * ESTRIDE;
        float se[16];
#pragma unroll
        for (int d = 0; d < 16; ++d) se[d] = 0.0f;
        float ssq = 0.0f;
        for (int f = 0; f < FEAT; ++f) {
#pragma unroll
            for (int d = 0; d < 16; ++d) {
                const float v = (float)er[f * 16 + d];
                se[d] += v; ssq += v * v;
            }
        }
        float s2 = 0.0f;
#pragma unroll
        for (int d = 0; d < 16; ++d) s2 += se[d] * se[d];
        out[(size_t)(row0 + tid) * OUTW] = fmf[tid] + bfp[0] + 0.5f * (s2 - ssq);
    }

    // ---------------- GEMM1: (64x1024 f16) @ W1^T -> relu -> H1sh ----------------
    const int rtp = (wv & 1) * 2;     // row tiles {rtp, rtp+1}
    const int cg  = wv >> 1;          // col tiles 4cg..4cg+3

    // per-thread staging constants (hoisted out of the k-loop)
    const _Float16* gsrc[4];          // async: global sources (+= kc*32 at use)
    int             lofs[4];          // async: LDS dest offsets within a Wbuf buffer
#pragma unroll
    for (int i = 0; i < 4; ++i) {
        const int s   = i * NTHR + tid;
        const int row = s >> 2;
        const int seg = (s & 3) * 8;  // halves
        gsrc[i] = wsW1 + (size_t)row * DNN_IN + seg;
        lofs[i] = row * WBSTRIDE + seg;
    }

    auto stage_w1 = [&](int kc, _Float16* dstbuf) {
        if constexpr (USE_ASYNC) {
#pragma unroll
            for (int i = 0; i < 4; ++i)
                async_ld_b128(gsrc[i] + kc * 32, dstbuf + lofs[i]);
        } else {
            const int row = tid;                           // 0..255
            const float* src = W1 + (size_t)row * DNN_IN + kc * 32;
            _Float16*    d   = dstbuf + row * WBSTRIDE;
#pragma unroll
            for (int i = 0; i < 2; ++i) {
                v4f a0 = *(const v4f*)(src + i * 16 + 0);
                v4f a1 = *(const v4f*)(src + i * 16 + 4);
                v4f a2 = *(const v4f*)(src + i * 16 + 8);
                v4f a3 = *(const v4f*)(src + i * 16 + 12);
                *(v8h*)(d + i * 16 + 0) = cvt8(a0, a1);
                *(v8h*)(d + i * 16 + 8) = cvt8(a2, a3);
            }
        }
    };

    v8f acc[2][4];
#pragma unroll
    for (int rr = 0; rr < 2; ++rr) {
#pragma unroll
        for (int t = 0; t < 4; ++t) {
#pragma unroll
            for (int j = 0; j < 8; ++j) acc[rr][t][j] = 0.0f;
        }
    }

    // hoisted fragment addressing: advance A pointers by 32 halves per k-step,
    // B offsets are loop-invariant within a buffer.
    const _Float16* a0p = Esh + (rtp * 16 + lm) * ESTRIDE + 16 * lhi;
    const _Float16* a1p = a0p + 16 * ESTRIDE;
    int bofs[4];
#pragma unroll
    for (int t = 0; t < 4; ++t) bofs[t] = (cg * 64 + t * 16 + lm) * WBSTRIDE + 16 * lhi;

    _Float16* cur = &Wbuf[0][0];
    _Float16* nxt = &Wbuf[1][0];

    stage_w1(0, cur);
    if constexpr (USE_ASYNC) wait_async0();
    __syncthreads();

#pragma unroll 1
    for (int kc = 0; kc < DNN_IN / 32; ++kc) {
        if (kc + 1 < DNN_IN / 32) stage_w1(kc + 1, nxt);
        const v16h af0 = lds_fragA(a0p);
        const v16h af1 = lds_fragA(a1p);
#pragma unroll
        for (int t = 0; t < 4; ++t) {
            const v16h bfr = lds_fragB(cur + bofs[t]);
            acc[0][t] = wmma_f16(af0, bfr, acc[0][t]);     // B-frag reused across 2 row tiles
            acc[1][t] = wmma_f16(af1, bfr, acc[1][t]);
        }
        if constexpr (USE_ASYNC) wait_async0();            // next chunk landed (ASYNCcnt)
        __syncthreads();                                   // all staged + consumed
        _Float16* tmp = cur; cur = nxt; nxt = tmp;         // parity-free double buffer
        a0p += 32; a1p += 32;
    }

    // epilogue: bias + relu -> H1sh.  C/D layout: VGPR j -> M = j + 8*lhi, lane lm -> N
#pragma unroll
    for (int rr = 0; rr < 2; ++rr) {
#pragma unroll
        for (int t = 0; t < 4; ++t) {
            const int n = cg * 64 + t * 16 + lm;
            const float bias = b1[n];
#pragma unroll
            for (int j = 0; j < 8; ++j) {
                float v = acc[rr][t][j] + bias;
                v = v > 0.0f ? v : 0.0f;
                H1sh[((rtp + rr) * 16 + 8 * lhi + j) * H1STRIDE + n] = (_Float16)v;
            }
        }
    }
    __syncthreads();

    // ---------------- GEMM2: (64x256 f16) @ W2^T -> relu -> out[:,1:129] ----------------
    const int ctp = (wv >> 1) * 2;    // col tiles {ctp, ctp+1} of 8
    v8f acc2[2][2];
#pragma unroll
    for (int rr = 0; rr < 2; ++rr) {
#pragma unroll
        for (int t = 0; t < 2; ++t) {
#pragma unroll
            for (int j = 0; j < 8; ++j) acc2[rr][t][j] = 0.0f;
        }
    }

    const _Float16* h0p = H1sh + (rtp * 16 + lm) * H1STRIDE + 16 * lhi;
    const _Float16* h1p = h0p + 16 * H1STRIDE;
    const _Float16* w2p0 = W2sh + ((ctp)     * 16 + lm) * W2STRIDE + 16 * lhi;
    const _Float16* w2p1 = W2sh + ((ctp + 1) * 16 + lm) * W2STRIDE + 16 * lhi;

#pragma unroll 2
    for (int kc = 0; kc < H1N / 32; ++kc) {
        const v16h af0 = lds_fragA(h0p + kc * 32);
        const v16h af1 = lds_fragA(h1p + kc * 32);
        {
            const v16h bfr = lds_fragB(w2p0 + kc * 32);
            acc2[0][0] = wmma_f16(af0, bfr, acc2[0][0]);
            acc2[1][0] = wmma_f16(af1, bfr, acc2[1][0]);
        }
        {
            const v16h bfr = lds_fragB(w2p1 + kc * 32);
            acc2[0][1] = wmma_f16(af0, bfr, acc2[0][1]);
            acc2[1][1] = wmma_f16(af1, bfr, acc2[1][1]);
        }
    }

#pragma unroll
    for (int rr = 0; rr < 2; ++rr) {
#pragma unroll
        for (int t = 0; t < 2; ++t) {
            const int n = (ctp + t) * 16 + lm;
            const float bias = b2[n];
#pragma unroll
            for (int j = 0; j < 8; ++j) {
                float v = acc2[rr][t][j] + bias;
                v = v > 0.0f ? v : 0.0f;
                out[(size_t)(row0 + (rtp + rr) * 16 + 8 * lhi + j) * OUTW + 1 + n] = v;
            }
        }
    }
}

extern "C" void kernel_launch(void* const* d_in, const int* in_sizes, int n_in,
                              void* d_out, int out_size, void* d_ws, size_t ws_size,
                              hipStream_t stream) {
    const float* x   = (const float*)d_in[0];
    const float* emb = (const float*)d_in[1];
    const float* Wc  = (const float*)d_in[2];
    const float* bc  = (const float*)d_in[3];
    const float* Wf  = (const float*)d_in[4];
    const float* bfp = (const float*)d_in[5];
    const float* W1  = (const float*)d_in[6];
    const float* b1  = (const float*)d_in[7];
    const float* W2  = (const float*)d_in[8];
    const float* b2  = (const float*)d_in[9];
    float* out = (float*)d_out;

    const int Bsz = in_sizes[0] / FEAT;                     // 65536
    const size_t w1h_bytes = (size_t)H1N * DNN_IN * sizeof(_Float16);  // 512 KB
    dim3 grid(Bsz / MT), block(NTHR);

    if (ws_size >= w1h_bytes) {
        _Float16* w1h = (_Float16*)d_ws;
        w1_to_f16<<<dim3((H1N * DNN_IN) / (256 * 4)), dim3(256), 0, stream>>>(W1, w1h);
        deepfm_fused<true><<<grid, block, 0, stream>>>(x, emb, Wc, bc, Wf, bfp,
                                                       W1, b1, W2, b2, w1h, out);
    } else {
        deepfm_fused<false><<<grid, block, 0, stream>>>(x, emb, Wc, bc, Wf, bfp,
                                                        W1, b1, W2, b2,
                                                        (const _Float16*)d_ws, out);
    }
}
